// DTAN_88759794139585
// MI455X (gfx1250) — matrix operand order repaired and checked by reading, not compiled, not run on hardware
//
#include <hip/hip_runtime.h>
#include <hip/hip_bf16.h>

typedef __attribute__((ext_vector_type(16))) _Float16 v16h;
typedef __attribute__((ext_vector_type(8)))  _Float16 h8;
typedef __attribute__((ext_vector_type(8)))  float    v8f;

// ---------------------------------------------------------------------------
// WMMA fragment layouts (CDNA5 ISA 7.12.2, wave32), f16 16x16x32:
//  A (MxK): lane -> M=lane&15; K chunks: elems 0-7 -> kbase+0..7,
//           elems 8-15 -> kbase+16..23, kbase = (lane>=16)?8:0
//           => two contiguous 8xf16 (16B) chunks, gap 16 elements.
//  B (KxN): lane -> N=lane&15; elems 0-7 -> koffB+0..7, 8-15 -> koffB+16..23, koffB=(lane>=16)?8:0
//           => two 8xf16 (16B) chunks, gap 16 elements, same as A.
//  C/D:     vgpr r -> M = r + ((lane>=16)?8:0), N = lane&15
// ---------------------------------------------------------------------------
__device__ __forceinline__ v16h load_fragA(const _Float16* p) {
  union { v16h v; h8 h[2]; } u;
  u.h[0] = *(const h8*)(p);        // K kbase+0..7
  u.h[1] = *(const h8*)(p + 16);   // K kbase+16..23
  return u.v;
}
__device__ __forceinline__ v16h load_fragB(const _Float16* p) {
  union { v16h v; h8 h[2]; } u;
  u.h[0] = *(const h8*)(p);        // K koffB+0..7
  u.h[1] = *(const h8*)(p + 16);   // K koffB+16..23
  return u.v;
}
#define WMMA_F16(A, B, C) \
  __builtin_amdgcn_wmma_f32_16x16x32_f16(false, (A), false, (B), (short)0, (C), false, false)

// ---------------------------------------------------------------------------
// Workspace layout (bytes, 256-aligned)
// ---------------------------------------------------------------------------
#define WS_W2T   0u            // [co][kc*128+ci] 64*1152 f16 = 147456
#define WS_W3T   147456u       // [co][kc*64+ci]  64*192  f16 =  24576
#define WS_FW1H  172032u       // [co][k] 16*2496 f16 =  79872
#define WS_X1P   251904u       // [n][p(264)][ci(128)] f16 = 1024*33792*2 = 69206016
#define WS_X2P   69457920u     // [n][p(96)][ci(64)]   f16 = 1024*6144*2  = 12582912
#define WS_XSH   82040832u     // [n][2496]            f16 = 5111808
#define WS_AB    87152640u     // [n][12]              f32 = 49152

// ---------------------------------------------------------------------------
// 0) Weight relayout f32 -> f16, K permuted to kc-major (kk = kc*C + ci)
// ---------------------------------------------------------------------------
__global__ void k_cvt_weights(const float* __restrict__ w2,
                              const float* __restrict__ w3,
                              const float* __restrict__ fw1,
                              _Float16* __restrict__ w2t,
                              _Float16* __restrict__ w3t,
                              _Float16* __restrict__ fw1h) {
  int i = blockIdx.x * 256 + threadIdx.x;
  if (i < 64 * 1152) {                 // w2 (64,128,9): w2t[co][kc*128+ci]
    int co = i / 1152, r = i - co * 1152;
    int kc = r >> 7, ci = r & 127;
    w2t[i] = (_Float16)w2[co * 1152 + ci * 9 + kc];
  }
  if (i < 64 * 192) {                  // w3 (64,64,3): w3t[co][kc*64+ci]
    int co = i / 192, r = i - co * 192;
    int kc = r >> 6, ci = r & 63;
    w3t[i] = (_Float16)w3[co * 192 + ci * 3 + kc];
  }
  if (i < 16 * 2496) {                 // fw1 already [co][k]
    fw1h[i] = (_Float16)fw1[i];
  }
}

// ---------------------------------------------------------------------------
// 1) conv1(7)+maxpool(3,2)+relu, C_in=1 stencil.
//    x (1024,1,512) -> x1p [n][p][ci], p padded 252->264 with zeros.
// ---------------------------------------------------------------------------
__global__ void k_conv1(const float* __restrict__ x,
                        const float* __restrict__ w1,
                        const float* __restrict__ b1,
                        _Float16* __restrict__ x1p) {
  __shared__ float sx[512];
  __shared__ float sw[128 * 7];
  __shared__ float sb[128];
  const int n = blockIdx.x, tid = threadIdx.x;
  for (int i = tid; i < 512; i += 256) sx[i] = x[n * 512 + i];
  for (int i = tid; i < 128 * 7; i += 256) sw[i] = w1[i];
  if (tid < 128) sb[tid] = b1[tid];
  __syncthreads();
  _Float16* xo = x1p + (size_t)n * 33792;
  for (int idx = tid; idx < 252 * 128; idx += 256) {
    int p = idx >> 7, c = idx & 127;
    const float* wr = sw + c * 7;
    float m = -1e30f;
    #pragma unroll
    for (int j = 0; j < 3; ++j) {
      float s = 0.0f;
      int base = 2 * p + j;
      #pragma unroll
      for (int t = 0; t < 7; ++t) s += sx[base + t] * wr[t];
      m = fmaxf(m, s);
    }
    xo[idx] = (_Float16)fmaxf(m + sb[c], 0.0f);
  }
  for (int idx = tid; idx < 12 * 128; idx += 256)   // zero pad rows 252..263
    xo[252 * 128 + idx] = (_Float16)0.0f;
}

// ---------------------------------------------------------------------------
// 2) conv2 WMMA GEMM: per batch M=244(pos) x N=64(co), K=1152 (36 k-tiles,
//    kk = kc*128+ci, kc = kt>>2, ci0 = (kt&3)*32). Each wave: 2 M-tiles x 4
//    N-tiles, B shared across M. Fused maxpool(3,3)+bias+relu -> x2p [p][ci].
// ---------------------------------------------------------------------------
__global__ void k_conv2(const _Float16* __restrict__ x1p,
                        const _Float16* __restrict__ w2t,
                        const float* __restrict__ b2,
                        _Float16* __restrict__ x2p) {
  extern __shared__ float ylds[];               // [co:64][pos:256] f32
  const int n = blockIdx.x, tid = threadIdx.x;
  const int lane = tid & 31, wave = tid >> 5;
  const int mrow  = lane & 15;
  const int kbase = (lane & 16) ? 8 : 0;
  const int koffB = (lane & 16) ? 8 : 0;
  const int mhi   = (lane & 16) ? 8 : 0;
  const _Float16* xb = x1p + (size_t)n * 33792;

  const int pA0 = wave * 16 + mrow;             // M-tile 0 row
  const int pA1 = pA0 + 128;                    // M-tile 1 row (mt+8)
  v8f acc[2][4] = {};
  for (int kt = 0; kt < 36; ++kt) {
    int kc  = kt >> 2;
    int cik = ((kt & 3) << 5) + kbase;
    v16h a0 = load_fragA(xb + (pA0 + kc) * 128 + cik);
    v16h a1 = load_fragA(xb + (pA1 + kc) * 128 + cik);
    const _Float16* wb = w2t + mrow * 1152 + kt * 32 + koffB;
    #pragma unroll
    for (int nt = 0; nt < 4; ++nt) {
      v16h b = load_fragB(wb + nt * (16 * 1152));
      acc[0][nt] = WMMA_F16(a0, b, acc[0][nt]);
      acc[1][nt] = WMMA_F16(a1, b, acc[1][nt]);
    }
  }
  #pragma unroll
  for (int h = 0; h < 2; ++h) {
    int p0 = (wave + 8 * h) * 16 + mhi;
    #pragma unroll
    for (int nt = 0; nt < 4; ++nt) {
      int co = nt * 16 + mrow;
      #pragma unroll
      for (int r = 0; r < 8; ++r)
        ylds[co * 256 + p0 + r] = acc[h][nt][r];
    }
  }
  __syncthreads();
  // maxpool(3,3)+bias+relu -> x2p [n][q][c], q padded 81->96
  _Float16* xo = x2p + (size_t)n * 6144;
  for (int idx = tid; idx < 81 * 64; idx += 256) {
    int q = idx >> 6, c = idx & 63;
    const float* row = ylds + c * 256 + q * 3;
    float m = fmaxf(fmaxf(row[0], row[1]), row[2]) + b2[c];
    xo[idx] = (_Float16)fmaxf(m, 0.0f);
  }
  for (int idx = tid; idx < 15 * 64; idx += 256)
    xo[81 * 64 + idx] = (_Float16)0.0f;
}

// ---------------------------------------------------------------------------
// 3) conv3 WMMA GEMM: per batch M=79(pos, 5 tiles) x N=64, K=192 (6 k-tiles,
//    kk = kc*64+ci, kc = kt>>1, ci0 = (kt&1)*32).
//    Fused maxpool(3,2)+bias+relu -> xs (1024,2496) with feature = c*39+q.
// ---------------------------------------------------------------------------
__global__ void k_conv3(const _Float16* __restrict__ x2p,
                        const _Float16* __restrict__ w3t,
                        const float* __restrict__ b3,
                        _Float16* __restrict__ xsh) {
  extern __shared__ float ylds[];               // [co:64][pos:80] f32
  const int n = blockIdx.x, tid = threadIdx.x;
  const int lane = tid & 31, wave = tid >> 5;
  const int mrow  = lane & 15;
  const int kbase = (lane & 16) ? 8 : 0;
  const int koffB = (lane & 16) ? 8 : 0;
  const int mhi   = (lane & 16) ? 8 : 0;
  const _Float16* xb = x2p + (size_t)n * 6144;

  for (int mt = wave; mt < 5; mt += 8) {
    const int p = mt * 16 + mrow;
    v8f acc[4] = {};
    for (int kt = 0; kt < 6; ++kt) {
      int kc  = kt >> 1;
      int cik = ((kt & 1) << 5) + kbase;
      v16h a = load_fragA(xb + (p + kc) * 64 + cik);
      const _Float16* wb = w3t + mrow * 192 + kt * 32 + koffB;
      #pragma unroll
      for (int nt = 0; nt < 4; ++nt) {
        v16h b = load_fragB(wb + nt * (16 * 192));
        acc[nt] = WMMA_F16(a, b, acc[nt]);
      }
    }
    int p0 = mt * 16 + mhi;
    #pragma unroll
    for (int nt = 0; nt < 4; ++nt) {
      int co = nt * 16 + mrow;
      #pragma unroll
      for (int r = 0; r < 8; ++r)
        ylds[co * 80 + p0 + r] = acc[nt][r];
    }
  }
  __syncthreads();
  _Float16* xo = xsh + (size_t)n * 2496;
  for (int idx = tid; idx < 64 * 39; idx += 256) {
    int c = idx / 39, q = idx - c * 39;
    const float* row = ylds + c * 80 + q * 2;
    float m = fmaxf(fmaxf(row[0], row[1]), row[2]) + b3[c];
    xo[idx] = (_Float16)fmaxf(m, 0.0f);     // feature = c*39+q
  }
}

// ---------------------------------------------------------------------------
// 4) FC1 (2496->16) WMMA (16 batches/block, N=16, 78 k-tiles, natural K),
//    fused bias+relu, FC2(16->5)+tanh, CPAB basis -> (a,b) per batch
// ---------------------------------------------------------------------------
__global__ void k_fc(const _Float16* __restrict__ xsh,
                     const _Float16* __restrict__ fw1h,
                     const float* __restrict__ fb1,
                     const float* __restrict__ fw2,
                     const float* __restrict__ fb2,
                     const float* __restrict__ Bb,   // (12,5) row-major
                     float* __restrict__ ab) {       // (1024,12): a[0..5], b[0..5]
  __shared__ float hlds[16 * 16];
  const int mt = blockIdx.x;
  const int lane = threadIdx.x & 31;
  const int mrow  = lane & 15;
  const int kbase = (lane & 16) ? 8 : 0;
  const int koffB = (lane & 16) ? 8 : 0;
  const int mhi   = (lane & 16) ? 8 : 0;
  const _Float16* xr = xsh + (size_t)(mt * 16 + mrow) * 2496;
  const _Float16* wr = fw1h + mrow * 2496;

  v8f acc = {};
  for (int kt = 0; kt < 78; ++kt) {
    v16h a = load_fragA(xr + kt * 32 + kbase);
    v16h b = load_fragB(wr + kt * 32 + koffB);
    acc = WMMA_F16(a, b, acc);
  }
  #pragma unroll
  for (int r = 0; r < 8; ++r)
    hlds[(r + mhi) * 16 + mrow] = fmaxf(acc[r] + fb1[mrow], 0.0f);
  __syncthreads();
  if (lane < 16) {
    int nrow = mt * 16 + lane;
    float th[5];
    #pragma unroll
    for (int j = 0; j < 5; ++j) {
      float s = fb2[j];
      #pragma unroll
      for (int i = 0; i < 16; ++i) s += hlds[lane * 16 + i] * fw2[j * 16 + i];
      th[j] = tanhf(s);
    }
    #pragma unroll
    for (int c = 0; c < 6; ++c) {
      float av = 0.0f, bv = 0.0f;
      #pragma unroll
      for (int j = 0; j < 5; ++j) {
        av += th[j] * Bb[(2 * c) * 5 + j];
        bv += th[j] * Bb[(2 * c + 1) * 5 + j];
      }
      ab[nrow * 12 + c]     = av;
      ab[nrow * 12 + 6 + c] = bv;
    }
  }
}

// ---------------------------------------------------------------------------
// 5) CPAB ODE integration (50 steps) + linear interpolation
// ---------------------------------------------------------------------------
__global__ void k_warp(const float* __restrict__ x,
                       const float* __restrict__ ab,
                       float* __restrict__ out) {
  __shared__ float s_ab[12];
  const int n = blockIdx.x, tid = threadIdx.x;
  if (tid < 12) s_ab[tid] = ab[n * 12 + tid];
  __syncthreads();
  const float dt = 1.0f / 50.0f;
  for (int i = tid; i < 512; i += 256) {
    float xx = (float)i * (1.0f / 511.0f);
    #pragma unroll 1
    for (int s = 0; s < 50; ++s) {
      int c = (int)(xx * 6.0f);
      c = c < 0 ? 0 : (c > 5 ? 5 : c);
      float ac = s_ab[c], bc = s_ab[6 + c];
      bool small = fabsf(ac) < 1e-6f;
      float sa = small ? 1.0f : ac;
      float e = expf(sa * dt);
      float xn = small ? (xx + bc * dt) : (e * xx + (bc / sa) * (e - 1.0f));
      xx = fminf(fmaxf(xn, 0.0f), 1.0f);
    }
    float pos = xx * 511.0f;
    int x0 = (int)floorf(pos);
    x0 = x0 < 0 ? 0 : (x0 > 510 ? 510 : x0);
    float w = pos - (float)x0;
    float g0 = x[n * 512 + x0];
    float g1 = x[n * 512 + x0 + 1];
    out[n * 512 + i] = g0 * (1.0f - w) + g1 * w;
  }
}

// ---------------------------------------------------------------------------
extern "C" void kernel_launch(void* const* d_in, const int* in_sizes, int n_in,
                              void* d_out, int out_size, void* d_ws, size_t ws_size,
                              hipStream_t stream) {
  const float* x   = (const float*)d_in[0];
  const float* w1  = (const float*)d_in[1];
  const float* b1  = (const float*)d_in[2];
  const float* w2  = (const float*)d_in[3];
  const float* b2  = (const float*)d_in[4];
  const float* w3  = (const float*)d_in[5];
  const float* b3  = (const float*)d_in[6];
  const float* fw1 = (const float*)d_in[7];
  const float* fb1 = (const float*)d_in[8];
  const float* fw2 = (const float*)d_in[9];
  const float* fb2 = (const float*)d_in[10];
  const float* Bb  = (const float*)d_in[11];

  char* ws = (char*)d_ws;
  _Float16* w2t  = (_Float16*)(ws + WS_W2T);
  _Float16* w3t  = (_Float16*)(ws + WS_W3T);
  _Float16* fw1h = (_Float16*)(ws + WS_FW1H);
  _Float16* x1p  = (_Float16*)(ws + WS_X1P);
  _Float16* x2p  = (_Float16*)(ws + WS_X2P);
  _Float16* xsh  = (_Float16*)(ws + WS_XSH);
  float*    abw  = (float*)   (ws + WS_AB);
  float*    out  = (float*)d_out;

  k_cvt_weights<<<(64 * 1152 + 255) / 256, 256, 0, stream>>>(w2, w3, fw1, w2t, w3t, fw1h);
  k_conv1<<<1024, 256, 0, stream>>>(x, w1, b1, x1p);
  k_conv2<<<1024, 256, 64 * 256 * sizeof(float), stream>>>(x1p, w2t, b2, x2p);
  k_conv3<<<1024, 256, 64 * 80 * sizeof(float), stream>>>(x2p, w3t, b3, xsh);
  k_fc<<<64, 32, 0, stream>>>(xsh, fw1h, fb1, fw2, fb2, Bb, abw);
  k_warp<<<1024, 256, 0, stream>>>(x, abw, out);
}
